// S2FM_50869592655332
// MI455X (gfx1250) — matrix operand order repaired and checked by reading, not compiled, not run on hardware
//
#include <hip/hip_runtime.h>
#include <hip/hip_bf16.h>

// ---------------------------------------------------------------------------
// MI455X / gfx1250 implementation.
// All heavy matmuls use v_wmma_f32_16x16x32_f16; 64x64x64 block tiles,
// register-pipelined global->LDS staging, 4 WMMAs per wave per K-stage.
// ---------------------------------------------------------------------------

typedef __attribute__((ext_vector_type(16))) _Float16 v16h;
typedef __attribute__((ext_vector_type(8)))  _Float16 v8h;
typedef __attribute__((ext_vector_type(8)))  float    v8f;

#define BN_EPS 1e-5f

__device__ __forceinline__ float sigf(float x) { return 1.f / (1.f + __expf(-x)); }

// Direction -> (h,w) for sequence position l (H=W=64).
__device__ __forceinline__ void dir_hw(int d, int l, int& h, int& w) {
  int q = l >> 6, r = l & 63;
  if (d == 0)      { h = q;      w = r;      }
  else if (d == 1) { h = q;      w = 63 - r; }
  else if (d == 2) { h = r;      w = q;      }
  else             { h = 63 - r; w = q;      }
}

// X4[m,k] gather from Fp (NCHW fp32, B=8, C=128, HW=4096), m in [0,131072)
__device__ __forceinline__ float x4_gather(const float* Fp, int m, int k) {
  int d = m >> 15, t = m & 32767;
  int b = t >> 12, l = t & 4095;
  int h, w; dir_hw(d, l, h, w);
  return Fp[(((size_t)(b * 128 + k)) << 12) + h * 64 + w];
}

__device__ __forceinline__ v16h frag16(v8h lo, v8h hi) {
  return __builtin_shufflevector(lo, hi, 0,1,2,3,4,5,6,7,8,9,10,11,12,13,14,15);
}

// ---------------------------------------------------------------------------
// Generic WMMA GEMM: C[M,N] = A[M,K] * B[K,N], f16 operands, f32 accum.
// Block: 256 threads = 8 waves; block tile 64(M) x 64(N); K step 64.
// Wave (wm in 0..1, wn in 0..3) owns two 16x16 C tiles sharing one B frag.
// Global fetch for stage k+1 is issued in registers while stage k computes.
// ---------------------------------------------------------------------------
template <typename P>
__global__ __launch_bounds__(256) void gemm_wmma(P p) {
  __shared__ __align__(16) _Float16 As[64][72];  // [m][k], padded rows
  __shared__ __align__(16) _Float16 Bs[64][72];  // [n][k] (transposed stage)

  const int tid  = threadIdx.x;
  const int lane = tid & 31;
  const int wave = tid >> 5;
  const int wm   = wave & 1;        // M half (rows wm*32 .. +31)
  const int wn   = wave >> 1;       // 0..3  (cols wn*16 .. +15)
  const int m0   = blockIdx.y * 64;
  const int n0   = blockIdx.x * 64;
  const int hs   = lane >> 4;       // half-select (lanes 0-15 vs 16-31)
  const int mn   = lane & 15;

  // Staging map: thread -> row sr (0..63), 16 contiguous k at sk.
  const int sr = tid >> 2;
  const int sk = (tid & 3) * 16;

  float ra[16], rb[16];
  auto fetch = [&](int kb) {
#pragma unroll
    for (int i = 0; i < 16; ++i) ra[i] = p.loadA(m0 + sr, kb + sk + i);
#pragma unroll
    for (int i = 0; i < 16; ++i) rb[i] = p.loadB(kb + sk + i, n0 + sr);
  };

  v8f acc0 = {}, acc1 = {};
  fetch(0);

  for (int kb = 0; kb < p.K; kb += 64) {
    __syncthreads();  // previous stage's fragment reads complete
#pragma unroll
    for (int i = 0; i < 16; ++i) As[sr][sk + i] = (_Float16)ra[i];
#pragma unroll
    for (int i = 0; i < 16; ++i) Bs[sr][sk + i] = (_Float16)rb[i];
    __syncthreads();

    // Issue next stage's global loads; latency overlaps the WMMAs below.
    if (kb + 64 < p.K) fetch(kb + 64);

#pragma unroll
    for (int kk = 0; kk < 64; kk += 32) {
      // B frag: lanes 0-15 hold K=kk+[0..7],[16..23]; lanes 16-31 hold K=kk+[8..15],[24..31]
      const _Float16* brow = &Bs[wn * 16 + mn][0];
      v16h b = frag16(*(const v8h*)&brow[kk + hs * 8],
                      *(const v8h*)&brow[kk + 16 + hs * 8]);
      // A frags per CDNA5 16-bit 16x32 layout:
      // lanes 0-15: K runs [0..7],[16..23]; lanes 16-31: [8..15],[24..31]
      const _Float16* ar0 = &As[wm * 32 + mn][0];
      v16h a0 = frag16(*(const v8h*)&ar0[kk + hs * 8],
                       *(const v8h*)&ar0[kk + 16 + hs * 8]);
      acc0 = __builtin_amdgcn_wmma_f32_16x16x32_f16(false, a0, false, b,
                                                    (short)0, acc0, false, false);
      const _Float16* ar1 = &As[wm * 32 + 16 + mn][0];
      v16h a1 = frag16(*(const v8h*)&ar1[kk + hs * 8],
                       *(const v8h*)&ar1[kk + 16 + hs * 8]);
      acc1 = __builtin_amdgcn_wmma_f32_16x16x32_f16(false, a1, false, b,
                                                    (short)0, acc1, false, false);
    }
  }

  // C layout: lane (hs,mn): N = n0+wn*16+mn; VGPR v -> M = v + hs*8
  const int n = n0 + wn * 16 + mn;
#pragma unroll
  for (int v = 0; v < 8; ++v) {
    p.store(m0 + wm * 32 + v + hs * 8, n, acc0[v]);
    p.store(m0 + wm * 32 + 16 + v + hs * 8, n, acc1[v]);
  }
}

// ---------------------------------------------------------------------------
// GEMM problem definitions (loaders fuse gathers / BN / activations)
// ---------------------------------------------------------------------------

// G1: x1 = relu(bn1(concat(FT1,FT2) @ primary_w))   M=32768 K=256 N=512
struct PG1 {
  int K;
  const float *FT1, *FT2, *W, *g_, *b_, *m_, *v_;
  _Float16* g;  // out: channels [0,512) of (B,1024,64,64)
  __device__ float loadA(int m, int k) const {
    int b = m >> 12, hw = m & 4095;
    const float* src = (k < 128) ? FT1 : FT2;
    return src[(((size_t)(b * 128 + (k & 127))) << 12) + hw];
  }
  __device__ float loadB(int k, int n) const { return W[n * 256 + k]; }
  __device__ void store(int m, int n, float v) const {
    int b = m >> 12, hw = m & 4095;
    float sc = g_[n] * rsqrtf(v_[n] + BN_EPS);
    float r = fmaxf((v - m_[n]) * sc + b_[n], 0.f);
    g[(((size_t)(b * 1024 + n)) << 12) + hw] = (_Float16)r;
  }
};

// G2: Fp = bn3((g * se) @ proj_w)                   M=32768 K=1024 N=128
struct PG2 {
  int K;
  const _Float16* g; const float* se; const float *W, *g_, *b_, *m_, *v_;
  float* Fp;
  __device__ float loadA(int m, int k) const {
    int b = m >> 12, hw = m & 4095;
    return (float)g[(((size_t)(b * 1024 + k)) << 12) + hw] * se[b * 1024 + k];
  }
  __device__ float loadB(int k, int n) const { return W[n * 1024 + k]; }
  __device__ void store(int m, int n, float v) const {
    int b = m >> 12, hw = m & 4095;
    float sc = g_[n] * rsqrtf(v_[n] + BN_EPS);
    Fp[(((size_t)(b * 128 + n)) << 12) + hw] = (v - m_[n]) * sc + b_[n];
  }
};

// G3: xz = X4 @ in_proj_w.T (X4 gathered from Fp)   M=131072 K=128 N=512
struct PG3 {
  int K;
  const float *Fp, *W;
  _Float16 *xs, *zb;  // each (131072, 256) row-major
  __device__ float loadA(int m, int k) const { return x4_gather(Fp, m, k); }
  __device__ float loadB(int k, int n) const { return W[n * 128 + k]; }
  __device__ void store(int m, int n, float v) const {
    if (n < 256) xs[(size_t)m * 256 + n] = (_Float16)v;
    else         zb[(size_t)m * 256 + (n - 256)] = (_Float16)v;
  }
};

// G4: Y4 = ys @ out_proj_w.T                        M=131072 K=256 N=128
struct PG4 {
  int K;
  const _Float16* ys; const float* W;
  _Float16* Y4;
  __device__ float loadA(int m, int k) const { return (float)ys[(size_t)m * 256 + k]; }
  __device__ float loadB(int k, int n) const { return W[n * 256 + k]; }
  __device__ void store(int m, int n, float v) const {
    Y4[(size_t)m * 128 + n] = (_Float16)v;
  }
};

// G5: pi = sigmoid([X4|Y4] @ pi_w.T + b); Yg = Y4*pi + X4*(1-pi)
//                                                   M=131072 K=256 N=128
struct PG5 {
  int K;
  const float* Fp; const _Float16* Y4; const float *W, *bias;
  _Float16* Yg;
  __device__ float loadA(int m, int k) const {
    if (k < 128) return x4_gather(Fp, m, k);
    return (float)Y4[(size_t)m * 128 + (k - 128)];
  }
  __device__ float loadB(int k, int n) const { return W[n * 256 + k]; }
  __device__ void store(int m, int n, float v) const {
    float pi = sigf(v + bias[n]);
    float x4 = x4_gather(Fp, m, n);
    float y4 = (float)Y4[(size_t)m * 128 + n];
    Yg[(size_t)m * 128 + n] = (_Float16)(y4 * pi + x4 * (1.f - pi));
  }
};

// G6: out = (f_scan + Fp) @ final_w.T               M=32768 K=128 N=128
struct PG6 {
  int K;
  const _Float16* fin; const float* W;
  float* out;  // (8,128,64,64) NCHW
  __device__ float loadA(int m, int k) const { return (float)fin[(size_t)m * 128 + k]; }
  __device__ float loadB(int k, int n) const { return W[n * 128 + k]; }
  __device__ void store(int m, int n, float v) const {
    int b = m >> 12, hw = m & 4095;
    out[(((size_t)(b * 128 + n)) << 12) + hw] = v;
  }
};

// ---------------------------------------------------------------------------
// Elementwise / small kernels
// ---------------------------------------------------------------------------

// Depthwise 3x3 + BN2 + ReLU. Reads g channels [0,512), writes [512,1024).
__global__ void k_dwconv(const float* cw, const float* g_, const float* b_,
                         const float* m_, const float* v_, _Float16* g) {
  int idx = blockIdx.x * blockDim.x + threadIdx.x;        // 8*512*4096
  int hw = idx & 4095, t = idx >> 12;
  int c = t & 511, b = t >> 9;
  int h = hw >> 6, w = hw & 63;
  float acc = 0.f;
#pragma unroll
  for (int i = 0; i < 3; ++i) {
#pragma unroll
    for (int j = 0; j < 3; ++j) {
      int hh = h + i - 1, ww = w + j - 1;
      if (hh >= 0 && hh < 64 && ww >= 0 && ww < 64)
        acc += (float)g[(((size_t)(b * 1024 + c)) << 12) + hh * 64 + ww] * cw[c * 9 + i * 3 + j];
    }
  }
  float sc = g_[c] * rsqrtf(v_[c] + BN_EPS);
  float r = fmaxf((acc - m_[c]) * sc + b_[c], 0.f);
  g[(((size_t)(b * 1024 + 512 + c)) << 12) + hw] = (_Float16)r;
}

// SE mean over HW: one block per (b,c) of 8*1024
__global__ void k_se_mean(const _Float16* g, float* s1) {
  __shared__ float red[256];
  int bc = blockIdx.x, tid = threadIdx.x;
  float s = 0.f;
#pragma unroll
  for (int i = 0; i < 16; ++i)
    s += (float)g[(((size_t)bc) << 12) + tid + 256 * i];
  red[tid] = s;
  __syncthreads();
  for (int o = 128; o > 0; o >>= 1) {
    if (tid < o) red[tid] += red[tid + o];
    __syncthreads();
  }
  if (tid == 0) s1[bc] = red[0] * (1.f / 4096.f);
}

// SE MLP: s = sigmoid(relu(s1 @ w1.T) @ w2.T). One block per batch (1024 thr).
__global__ void k_se_mlp(const float* s1, const float* w1, const float* w2, float* se) {
  __shared__ float hid[64];
  int b = blockIdx.x, tid = threadIdx.x;
  if (tid < 64) {
    float a = 0.f;
    for (int k = 0; k < 1024; ++k) a += s1[b * 1024 + k] * w1[tid * 1024 + k];
    hid[tid] = fmaxf(a, 0.f);
  }
  __syncthreads();
  float a = 0.f;
#pragma unroll
  for (int j = 0; j < 64; ++j) a += hid[j] * w2[tid * 64 + j];
  se[b * 1024 + tid] = sigf(a);
}

// Causal depthwise conv1d (K=4) + SiLU: u from xs.
__global__ void k_conv1d(const _Float16* xs, const float* w, const float* bias,
                         _Float16* ub) {
  int idx = blockIdx.x * blockDim.x + threadIdx.x;   // 32*4096*256
  int c = idx & 255, p = idx >> 8;
  int l = p & 4095, seq = p >> 12;
  float acc = bias[c];
#pragma unroll
  for (int t = 0; t < 4; ++t) {
    int ls = l - 3 + t;
    if (ls >= 0)
      acc += (float)xs[((size_t)(seq * 4096 + ls)) * 256 + c] * w[c * 4 + t];
  }
  ub[(size_t)p * 256 + c] = (_Float16)(acc * sigf(acc));
}

// dbl = u @ x_proj_w.T  (10 outputs per position)
__global__ void k_xproj(const _Float16* ub, const float* xw, float* dbl) {
  int idx = blockIdx.x * blockDim.x + threadIdx.x;   // 131072*16
  int j = idx & 15, p = idx >> 4;
  if (j >= 10) return;
  float a = 0.f;
  for (int k = 0; k < 256; ++k) a += (float)ub[(size_t)p * 256 + k] * xw[j * 256 + k];
  dbl[(size_t)p * 10 + j] = a;
}

// dt = softplus(dbl[:8] @ dt_proj_w.T + b)
__global__ void k_dt(const float* dbl, const float* dtw, const float* dtb_b,
                     _Float16* dtb) {
  int idx = blockIdx.x * blockDim.x + threadIdx.x;   // 131072*256
  int c = idx & 255, p = idx >> 8;
  float a = dtb_b[c];
#pragma unroll
  for (int j = 0; j < 8; ++j) a += dbl[(size_t)p * 10 + j] * dtw[c * 8 + j];
  float sp = (a > 20.f) ? a : log1pf(__expf(a));
  dtb[(size_t)p * 256 + c] = (_Float16)sp;
}

// Selective scan (DST=1): one block per sequence, one thread per channel.
// Fused epilogue: y = (h*C + u*Dp) * silu(z) -> ys (f16).
__global__ void k_scan(const _Float16* dtb, const _Float16* ub, const _Float16* zb,
                       const float* dbl, const float* A_log, const float* Dp,
                       _Float16* ys) {
  int seq = blockIdx.x;       // 0..31
  int c = threadIdx.x;        // 0..255
  float A = -__expf(A_log[c]);
  float D = Dp[c];
  float h = 0.f;
  size_t base = (size_t)seq * 4096;
  for (int l = 0; l < 4096; ++l) {
    size_t p = base + l;
    float dt = (float)dtb[p * 256 + c];
    float u  = (float)ub[p * 256 + c];
    float Bv = dbl[p * 10 + 8];
    float Cv = dbl[p * 10 + 9];
    h = h * __expf(dt * A) + dt * Bv * u;
    float y = h * Cv + u * D;
    float z = (float)zb[p * 256 + c];
    ys[p * 256 + c] = (_Float16)(y * (z * sigf(z)));
  }
}

// Inverse-permute sum over the 4 directions + residual Fp -> fin (f16, MK).
__global__ void k_reduce4(const _Float16* Yg, const float* Fp, _Float16* fin) {
  int idx = blockIdx.x * blockDim.x + threadIdx.x;   // 8*4096*128
  int c = idx & 127, pos = idx >> 7;
  int b = pos >> 12, hw = pos & 4095;
  int h = hw >> 6, w = hw & 63;
  int l0 = hw;
  int l1 = (h << 6) + (63 - w);
  int l2 = (w << 6) + h;
  int l3 = (w << 6) + (63 - h);
  float s = Fp[(((size_t)(b * 128 + c)) << 12) + hw];
  s += (float)Yg[(((size_t)((0 * 8 + b)) << 12) + l0) * 128 + c];
  s += (float)Yg[(((size_t)((1 * 8 + b)) << 12) + l1) * 128 + c];
  s += (float)Yg[(((size_t)((2 * 8 + b)) << 12) + l2) * 128 + c];
  s += (float)Yg[(((size_t)((3 * 8 + b)) << 12) + l3) * 128 + c];
  fin[((size_t)(b * 4096 + hw)) * 128 + c] = (_Float16)s;
}

// ---------------------------------------------------------------------------
// Launch
// ---------------------------------------------------------------------------
extern "C" void kernel_launch(void* const* d_in, const int* in_sizes, int n_in,
                              void* d_out, int out_size, void* d_ws, size_t ws_size,
                              hipStream_t stream) {
  const float* FT1       = (const float*)d_in[0];
  const float* FT2       = (const float*)d_in[1];
  const float* primary_w = (const float*)d_in[2];
  const float* bn1_g     = (const float*)d_in[3];
  const float* bn1_b     = (const float*)d_in[4];
  const float* bn1_m     = (const float*)d_in[5];
  const float* bn1_v     = (const float*)d_in[6];
  const float* cheap_w   = (const float*)d_in[7];
  const float* bn2_g     = (const float*)d_in[8];
  const float* bn2_b     = (const float*)d_in[9];
  const float* bn2_m     = (const float*)d_in[10];
  const float* bn2_v     = (const float*)d_in[11];
  const float* se_w1     = (const float*)d_in[12];
  const float* se_w2     = (const float*)d_in[13];
  const float* proj_w    = (const float*)d_in[14];
  const float* bn3_g     = (const float*)d_in[15];
  const float* bn3_b     = (const float*)d_in[16];
  const float* bn3_m     = (const float*)d_in[17];
  const float* bn3_v     = (const float*)d_in[18];
  const float* in_proj_w = (const float*)d_in[19];
  const float* conv1d_w  = (const float*)d_in[20];
  const float* conv1d_b  = (const float*)d_in[21];
  const float* x_proj_w  = (const float*)d_in[22];
  const float* dt_proj_w = (const float*)d_in[23];
  const float* dt_proj_b = (const float*)d_in[24];
  const float* A_log     = (const float*)d_in[25];
  const float* Dp        = (const float*)d_in[26];
  const float* out_proj_w= (const float*)d_in[27];
  const float* pi_w      = (const float*)d_in[28];
  const float* pi_b      = (const float*)d_in[29];
  const float* final_w   = (const float*)d_in[30];
  float* out = (float*)d_out;

  // Workspace carve-out (256B aligned)
  char* ws = (char*)d_ws;
  size_t off = 0;
  auto carve = [&](size_t bytes) -> void* {
    void* p = ws + off;
    off += (bytes + 255) & ~(size_t)255;
    return p;
  };
  const size_t M1 = 32768, M4 = 131072;
  _Float16* g   = (_Float16*)carve(8ull * 1024 * 4096 * 2);   // ghost feature (f16)
  float*    Fp  = (float*)   carve(8ull * 128 * 4096 * 4);
  float*    s1  = (float*)   carve(8 * 1024 * 4);
  float*    se  = (float*)   carve(8 * 1024 * 4);
  _Float16* xs  = (_Float16*)carve(M4 * 256 * 2);
  _Float16* zb  = (_Float16*)carve(M4 * 256 * 2);
  _Float16* ub  = (_Float16*)carve(M4 * 256 * 2);
  float*    dbl = (float*)   carve(M4 * 10 * 4);
  _Float16* dtb = (_Float16*)carve(M4 * 256 * 2);
  _Float16* Y4  = (_Float16*)carve(M4 * 128 * 2);
  _Float16* ys  = xs;    // xs dead after conv1d
  _Float16* Yg  = dtb;   // dtb dead after scan
  _Float16* fin = ub;    // ub dead after scan

  // G1: primary 1x1 conv + BN1 + ReLU  -> g[:,0:512]
  {
    PG1 p{256, FT1, FT2, primary_w, bn1_g, bn1_b, bn1_m, bn1_v, g};
    gemm_wmma<PG1><<<dim3(512 / 64, M1 / 64), 256, 0, stream>>>(p);
  }
  // depthwise 3x3 + BN2 + ReLU -> g[:,512:1024]
  k_dwconv<<<(8 * 512 * 4096) / 256, 256, 0, stream>>>(cheap_w, bn2_g, bn2_b, bn2_m, bn2_v, g);
  // SE
  k_se_mean<<<8 * 1024, 256, 0, stream>>>(g, s1);
  k_se_mlp<<<8, 1024, 0, stream>>>(s1, se_w1, se_w2, se);
  // G2: projection + BN3 -> Fp
  {
    PG2 p{1024, g, se, proj_w, bn3_g, bn3_b, bn3_m, bn3_v, Fp};
    gemm_wmma<PG2><<<dim3(128 / 64, M1 / 64), 256, 0, stream>>>(p);
  }
  // G3: in_proj (X4 gathered from Fp) -> xs, z
  {
    PG3 p{128, Fp, in_proj_w, xs, zb};
    gemm_wmma<PG3><<<dim3(512 / 64, M4 / 64), 256, 0, stream>>>(p);
  }
  // conv1d + silu -> u
  k_conv1d<<<(32 * 4096 * 256) / 256, 256, 0, stream>>>(xs, conv1d_w, conv1d_b, ub);
  // x_proj -> dbl (dt_raw | B | C)
  k_xproj<<<(int)(M4 * 16 / 256), 256, 0, stream>>>(ub, x_proj_w, dbl);
  // dt projection + softplus
  k_dt<<<(int)(M4 * 256 / 256), 256, 0, stream>>>(dbl, dt_proj_w, dt_proj_b, dtb);
  // selective scan + gating epilogue -> ys
  k_scan<<<32, 256, 0, stream>>>(dtb, ub, zb, dbl, A_log, Dp, ys);
  // G4: out_proj -> Y4
  {
    PG4 p{256, ys, out_proj_w, Y4};
    gemm_wmma<PG4><<<dim3(128 / 64, M4 / 64), 256, 0, stream>>>(p);
  }
  // G5: pi gate + blend -> Yg
  {
    PG5 p{256, Fp, Y4, pi_w, pi_b, Yg};
    gemm_wmma<PG5><<<dim3(128 / 64, M4 / 64), 256, 0, stream>>>(p);
  }
  // inverse-permute 4-direction sum + Fp residual -> fin
  k_reduce4<<<(8 * 4096 * 128) / 256, 256, 0, stream>>>(Yg, Fp, fin);
  // G6: final 1x1 conv -> out
  {
    PG6 p{128, fin, final_w, out};
    gemm_wmma<PG6><<<dim3(128 / 64, M1 / 64), 256, 0, stream>>>(p);
  }
  (void)in_sizes; (void)n_in; (void)out_size; (void)ws_size;
}
